// DGNN_90357521973791
// MI455X (gfx1250) — compile-verified, matrix operand order checked by reading, not run on hardware
//
#include <hip/hip_runtime.h>
#include <math.h>

// ---------------------------------------------------------------------------
// CDNA5 (gfx1250) DGNN implementation.
// Heavy matmuls: v_wmma_f32_16x16x32_bf16, fp32 accumulate.
//  - 128x256x32 block tile, 8 waves, 64x64 per wave (16 WMMA / 16 ds_b128)
//  - direct global->bf16->LDS staging, double-buffered LDS, 1 barrier/chunk
//  - rolling segment state (no per-chunk pointer selects)
//  - segmented-K GEMM: temp11 = sum_i alpha_i * Xi @ Xi^T in a single pass
//  - fused epilogues: accumulate / (aux - s*sigmoid(acc)) / bias
// ---------------------------------------------------------------------------

typedef __attribute__((ext_vector_type(16))) __bf16          v16bf;
typedef __attribute__((ext_vector_type(8)))  float           v8f;
typedef __attribute__((ext_vector_type(16))) unsigned short  v16u;
typedef __attribute__((ext_vector_type(8)))  unsigned short  v8u;

__device__ __forceinline__ unsigned short f2bfu(float x) {
  return __builtin_bit_cast(unsigned short, (__bf16)x);
}
// pack two f32 -> two bf16 in one dword (clang lowers __bf16 casts to HW cvt)
__device__ __forceinline__ unsigned pk2bf(float x, float y) {
  return (unsigned)f2bfu(x) | ((unsigned)f2bfu(y) << 16);
}

#define BM   128
#define BN   256
#define BK   32
#define LSTR 40   // padded LDS row stride (80B) -> conflict-free b128 frag reads

// Epilogue modes:
//  0: C = acc (+ C if accflag)
//  1: C = aux - s * sigmoid(acc)        (aux same shape/ld as C)
//  2: C = acc + aux[col]                (bias)
// A is staged as av_seg * A_seg (per-K-segment pointer and scale).
template <bool TRANSB>
__global__ __launch_bounds__(256, 2)
void gemm_wmma_bf16(
    const float* __restrict__ A0, const float* __restrict__ A1, const float* __restrict__ A2,
    const float* __restrict__ B0, const float* __restrict__ B1, const float* __restrict__ B2,
    float* __restrict__ C, const float* __restrict__ aux,
    int K, int segLen, int lda, int ldb, int ldc,
    float av0, float av1, float av2, float s, int mode, int accflag) {
  __shared__ unsigned short As[2][BM * LSTR];
  __shared__ unsigned short Bs[2][BN * LSTR];

  const int tid  = threadIdx.x;
  const int lane = tid & 31;
  const int wv   = tid >> 5;        // 8 waves: 2 (m) x 4 (n)
  const int wm   = wv >> 2;         // 0..1 -> 64-row slab
  const int wn   = wv & 3;          // 0..3 -> 64-col slab
  const int m0   = blockIdx.y * BM;
  const int n0   = blockIdx.x * BN;

  v8f acc[4][4];
#pragma unroll
  for (int mt = 0; mt < 4; ++mt)
#pragma unroll
    for (int nt = 0; nt < 4; ++nt)
#pragma unroll
      for (int e = 0; e < 8; ++e) acc[mt][nt][e] = 0.f;

  // global load -> scale -> convert/pack -> LDS store
  auto stage = [&](int buf, const float* __restrict__ Ap, const float* __restrict__ Bp,
                   float av, int kk) {
#pragma unroll
    for (int i = 0; i < 4; ++i) {
      const int idx = tid + i * 256;
      const int r = idx >> 3, c4 = (idx & 7) << 2;
      float4 v = *(const float4*)(Ap + (size_t)(m0 + r) * lda + kk + c4);
      if (av != 1.0f) { v.x *= av; v.y *= av; v.z *= av; v.w *= av; }
      uint2 pk; pk.x = pk2bf(v.x, v.y); pk.y = pk2bf(v.z, v.w);
      *(uint2*)(&As[buf][r * LSTR + c4]) = pk;        // ds_store_b64
    }
    if (TRANSB) {  // B is [N,K] row-major
#pragma unroll
      for (int i = 0; i < 8; ++i) {
        const int idx = tid + i * 256;
        const int r = idx >> 3, c4 = (idx & 7) << 2;
        const float4 v = *(const float4*)(Bp + (size_t)(n0 + r) * ldb + kk + c4);
        uint2 pk; pk.x = pk2bf(v.x, v.y); pk.y = pk2bf(v.z, v.w);
        *(uint2*)(&Bs[buf][r * LSTR + c4]) = pk;      // ds_store_b64
      }
    } else {       // B is [K,N] row-major: transpose-on-store into Bs[n][k]
#pragma unroll
      for (int i = 0; i < 8; ++i) {
        const int idx = tid + i * 256;
        const int k = idx >> 6, c4 = (idx & 63) << 2;
        const float4 v = *(const float4*)(Bp + (size_t)(kk + k) * ldb + n0 + c4);
        Bs[buf][(c4 + 0) * LSTR + k] = f2bfu(v.x);
        Bs[buf][(c4 + 1) * LSTR + k] = f2bfu(v.y);
        Bs[buf][(c4 + 2) * LSTR + k] = f2bfu(v.z);
        Bs[buf][(c4 + 3) * LSTR + k] = f2bfu(v.w);
      }
    }
  };

  // rolling segment state: updated only on (rare, uniform) segment rollover
  const float* Ap = A0;
  const float* Bp = B0;
  float av = av0;
  int   kk = 0;
  int   segi = 0;

  stage(0, Ap, Bp, av, 0);
  __syncthreads();

  // ISA 16-bit fragment layout: lanes 0-15 -> K {0..7,16..23}, lanes 16-31 -> K {8..15,24..31}
  const int kb0  = (lane < 16) ? 0 : 8;
  const int rsel = lane & 15;
  int cur = 0;

  for (int kb = 0; kb < K; kb += BK) {
    v16bf bfr[4];
#pragma unroll
    for (int nt = 0; nt < 4; ++nt) {
      const unsigned short* p = &Bs[cur][(wn * 64 + nt * 16 + rsel) * LSTR + kb0];
      const v8u lo = *(const v8u*)p;
      const v8u hi = *(const v8u*)(p + 16);
      v16u t;
#pragma unroll
      for (int e = 0; e < 8; ++e) { t[e] = lo[e]; t[e + 8] = hi[e]; }
      bfr[nt] = __builtin_bit_cast(v16bf, t);
    }
#pragma unroll
    for (int mt = 0; mt < 4; ++mt) {
      const unsigned short* p = &As[cur][(wm * 64 + mt * 16 + rsel) * LSTR + kb0];
      const v8u lo = *(const v8u*)p;
      const v8u hi = *(const v8u*)(p + 16);
      v16u t;
#pragma unroll
      for (int e = 0; e < 8; ++e) { t[e] = lo[e]; t[e + 8] = hi[e]; }
      const v16bf afr = __builtin_bit_cast(v16bf, t);
#pragma unroll
      for (int nt = 0; nt < 4; ++nt)
        acc[mt][nt] = __builtin_amdgcn_wmma_f32_16x16x32_bf16(
            false, afr, false, bfr[nt], (short)0, acc[mt][nt], false, false);
    }

    if (kb + BK < K) {
      kk += BK;
      if (kk == segLen) {           // rare, uniform: advance to next K segment
        kk = 0;
        ++segi;
        Ap = (segi == 1) ? A1 : A2;
        Bp = (segi == 1) ? B1 : B2;
        av = (segi == 1) ? av1 : av2;
      }
      stage(cur ^ 1, Ap, Bp, av, kk);   // fill the other LDS buffer
      __syncthreads();
      cur ^= 1;
    }
  }

  // ---- epilogue: C/D layout: col = lane%16, row = v + 8*(lane/16) ----
  const int cn = lane & 15;
  const int rh = (lane >> 4) << 3;

#define EPI_LOOP(BODY)                                               \
  _Pragma("unroll") for (int mt = 0; mt < 4; ++mt)                   \
  _Pragma("unroll") for (int nt = 0; nt < 4; ++nt) {                 \
    const int col  = n0 + wn * 64 + nt * 16 + cn;                    \
    const int rowb = m0 + wm * 64 + mt * 16 + rh;                    \
    _Pragma("unroll") for (int v = 0; v < 8; ++v) {                  \
      const size_t off = (size_t)(rowb + v) * ldc + col;             \
      const float  x   = acc[mt][nt][v];                             \
      BODY                                                           \
    }                                                                \
  }

  if (mode == 1) {
    EPI_LOOP(C[off] = aux[off] - s * (1.0f / (1.0f + __expf(-x)));)
  } else if (mode == 2) {
    EPI_LOOP(C[off] = x + aux[col];)
  } else if (accflag) {
    EPI_LOOP(C[off] = x + C[off];)
  } else {
    EPI_LOOP(C[off] = x;)
  }
#undef EPI_LOOP
}

// row-wise L2 normalize x[N,F] into p[:, coloff:coloff+F] (ldp = 3F)
__global__ __launch_bounds__(256) void l2norm_concat(const float* __restrict__ x,
                                                     float* __restrict__ p,
                                                     int F, int ldp, int coloff) {
  __shared__ float red[256];
  const int row = blockIdx.x;
  const int t   = threadIdx.x;
  float s = 0.f;
  for (int c = t; c < F; c += 256) { const float v = x[(size_t)row * F + c]; s += v * v; }
  red[t] = s; __syncthreads();
  for (int st = 128; st > 0; st >>= 1) { if (t < st) red[t] += red[t + st]; __syncthreads(); }
  const float inv = 1.0f / fmaxf(sqrtf(red[0]), 1e-12f);
  for (int c = t; c < F; c += 256)
    p[(size_t)row * ldp + coloff + c] = x[(size_t)row * F + c] * inv;
}

__global__ __launch_bounds__(256) void log_softmax_rows(const float* __restrict__ in,
                                                        float* __restrict__ out, int F) {
  __shared__ float red[256];
  const int row = blockIdx.x;
  const int t   = threadIdx.x;
  float m = -3.402823e38f;
  for (int c = t; c < F; c += 256) m = fmaxf(m, in[(size_t)row * F + c]);
  red[t] = m; __syncthreads();
  for (int st = 128; st > 0; st >>= 1) { if (t < st) red[t] = fmaxf(red[t], red[t + st]); __syncthreads(); }
  const float mx = red[0]; __syncthreads();
  float s = 0.f;
  for (int c = t; c < F; c += 256) s += __expf(in[(size_t)row * F + c] - mx);
  red[t] = s; __syncthreads();
  for (int st = 128; st > 0; st >>= 1) { if (t < st) red[t] += red[t + st]; __syncthreads(); }
  const float lse = mx + __logf(red[0]);
  for (int c = t; c < F; c += 256) out[(size_t)row * F + c] = in[(size_t)row * F + c] - lse;
}

static inline void gemm(hipStream_t st, bool transB, float* C, const float* A,
                        const float* B, const float* aux, int M, int N, int K,
                        int lda, int ldb, int ldc, float av, float s, int mode,
                        int accflag) {
  dim3 g(N / BN, M / BM), b(256);
  if (transB)
    gemm_wmma_bf16<true><<<g, b, 0, st>>>(A, A, A, B, B, B, C, aux, K, K,
                                          lda, ldb, ldc, av, av, av, s, mode, accflag);
  else
    gemm_wmma_bf16<false><<<g, b, 0, st>>>(A, A, A, B, B, B, C, aux, K, K,
                                           lda, ldb, ldc, av, av, av, s, mode, accflag);
}

// C = a0*X0@X0^T + a1*X1@X1^T + a2*X2@X2^T   (single pass over C)
static inline void gram3(hipStream_t st, float* C, const float* X0, const float* X1,
                         const float* X2, int M, int N, int K1, float a0, float a1,
                         float a2, int ldx, int ldc) {
  dim3 g(N / BN, M / BM), b(256);
  gemm_wmma_bf16<true><<<g, b, 0, st>>>(X0, X1, X2, X0, X1, X2, C, nullptr,
                                        3 * K1, K1, ldx, ldx, ldc, a0, a1, a2,
                                        0.f, 0, 0);
}

extern "C" void kernel_launch(void* const* d_in, const int* in_sizes, int n_in,
                              void* d_out, int out_size, void* d_ws, size_t ws_size,
                              hipStream_t stream) {
  (void)in_sizes; (void)n_in; (void)out_size; (void)ws_size;
  constexpr int Nn = 4096, Ff = 512, F3 = 1536;
  const float* feat = (const float*)d_in[0];
  const float* adj  = (const float*)d_in[1];
  const float* adj1 = (const float*)d_in[2];
  const float* y_in = (const float*)d_in[3];
  const float* wy   = (const float*)d_in[4];
  const float* w2w  = (const float*)d_in[5];   // [F, 3F]
  const float* w2b  = (const float*)d_in[6];   // [F]
  float* outp = (float*)d_out;

  const size_t NF = (size_t)Nn * Ff;
  float* w = (float*)d_ws;
  float* ya  = w; w += NF;  float* z1a = w; w += NF;  float* z2a = w; w += NF;
  float* yb  = w; w += NF;  float* z1b = w; w += NF;  float* z2b = w; w += NF;
  float* ywb = w; w += NF;  float* z1w = w; w += NF;  float* z2w = w; w += NF;
  float* t12 = w; w += NF;  float* t24 = w; w += NF;  float* t34 = w; w += NF;
  float* xT  = w; w += NF;  float* az  = w; w += NF;
  float* p   = w; w += 3 * NF;
  float* lg  = w; w += NF;
  float* g11 = w; w += (size_t)Nn * Nn;          // temp11 [N,N]

  const float EPS = 0.5f, LAM1 = 1.0f, LAM2 = 1.0f, LAM3 = 0.5f;

  const float* yc  = y_in;
  const float* z1c = feat;
  const float* z2c = feat;
  float* yn = ya; float* z1n = z1a; float* z2n = z2a;

  for (int layer = 0; layer < 2; ++layer) {
    // xw = x @ wy
    gemm(stream, false, ywb, yc,  wy, nullptr, Nn, Ff, Ff, Ff, Ff, Ff, 1.f, 0.f, 0, 0);
    gemm(stream, false, z1w, z1c, wy, nullptr, Nn, Ff, Ff, Ff, Ff, Ff, 1.f, 0.f, 0, 0);
    gemm(stream, false, z2w, z2c, wy, nullptr, Nn, Ff, Ff, Ff, Ff, Ff, 1.f, 0.f, 0, 0);

    // temp11 = yw@yw.T - EPS*z1w@z1w.T - (1-EPS)*z2w@z2w.T  (one fused pass)
    gram3(stream, g11, ywb, z1w, z2w, Nn, Nn, Ff, 1.f, -EPS, -(1.f - EPS), Ff, Nn);

    // tX = xw@wy.T + (x@wy.T)@wy
    gemm(stream, true,  t12, ywb, wy, nullptr, Nn, Ff, Ff, Ff, Ff, Ff, 1.f, 0.f, 0, 0);
    gemm(stream, true,  xT,  yc,  wy, nullptr, Nn, Ff, Ff, Ff, Ff, Ff, 1.f, 0.f, 0, 0);
    gemm(stream, false, t12, xT,  wy, nullptr, Nn, Ff, Ff, Ff, Ff, Ff, 1.f, 0.f, 0, 1);

    gemm(stream, true,  t24, z1w, wy, nullptr, Nn, Ff, Ff, Ff, Ff, Ff, 1.f, 0.f, 0, 0);
    gemm(stream, true,  xT,  z1c, wy, nullptr, Nn, Ff, Ff, Ff, Ff, Ff, 1.f, 0.f, 0, 0);
    gemm(stream, false, t24, xT,  wy, nullptr, Nn, Ff, Ff, Ff, Ff, Ff, 1.f, 0.f, 0, 1);

    gemm(stream, true,  t34, z2w, wy, nullptr, Nn, Ff, Ff, Ff, Ff, Ff, 1.f, 0.f, 0, 0);
    gemm(stream, true,  xT,  z2c, wy, nullptr, Nn, Ff, Ff, Ff, Ff, Ff, 1.f, 0.f, 0, 0);
    gemm(stream, false, t34, xT,  wy, nullptr, Nn, Ff, Ff, Ff, Ff, Ff, 1.f, 0.f, 0, 1);

    // y_next = feat - LAM3 * sigmoid(+temp11 @ t12)
    gemm(stream, false, yn, g11, t12, feat, Nn, Ff, Nn, Nn, Ff, Ff, 1.f, LAM3, 1, 0);

    // z1_next = adj@z1 - EPS*(LAM3/LAM1) * sigmoid(-temp11 @ t24)   (A scaled by -1)
    gemm(stream, false, az,  adj, z1c, nullptr, Nn, Ff, Nn, Nn, Ff, Ff, 1.f, 0.f, 0, 0);
    gemm(stream, false, z1n, g11, t24, az, Nn, Ff, Nn, Nn, Ff, Ff, -1.f, EPS * LAM3 / LAM1, 1, 0);

    // z2_next = adj1@z2 - (1-EPS)*(LAM3/LAM2) * sigmoid(-temp11 @ t34)
    gemm(stream, false, az,  adj1, z2c, nullptr, Nn, Ff, Nn, Nn, Ff, Ff, 1.f, 0.f, 0, 0);
    gemm(stream, false, z2n, g11, t34, az, Nn, Ff, Nn, Nn, Ff, Ff, -1.f, (1.f - EPS) * LAM3 / LAM2, 1, 0);

    yc = yn; z1c = z1n; z2c = z2n;
    yn = yb; z1n = z1b; z2n = z2b;
  }

  // p = [l2norm(y) | l2norm(z1) | l2norm(z2)]
  l2norm_concat<<<Nn, 256, 0, stream>>>(yc,  p, Ff, F3, 0);
  l2norm_concat<<<Nn, 256, 0, stream>>>(z1c, p, Ff, F3, Ff);
  l2norm_concat<<<Nn, 256, 0, stream>>>(z2c, p, Ff, F3, 2 * Ff);

  // logits = p @ w2_w.T + w2_b
  gemm(stream, true, lg, p, w2w, w2b, Nn, Ff, F3, F3, F3, Ff, 1.f, 0.f, 2, 0);

  log_softmax_rows<<<Nn, 256, 0, stream>>>(lg, outp, Ff);
}
